// GatEncoder_87952340288231
// MI455X (gfx1250) — matrix operand order repaired and checked by reading, not compiled, not run on hardware
//
#include <hip/hip_runtime.h>
#include <hip/hip_bf16.h>

// ---------------------------------------------------------------------------
// GAT encoder for MI455X (gfx1250): bf16 WMMA (f32 accum), fused softmax GEMM,
// TDM tensor_load_to_lds for the attention B-tiles.
// ---------------------------------------------------------------------------

typedef __attribute__((ext_vector_type(16))) __bf16    v16bf;
typedef __attribute__((ext_vector_type(8)))  float     v8f;
typedef __attribute__((ext_vector_type(4)))  unsigned  v4u;
typedef __attribute__((ext_vector_type(8)))  int       v8i;
typedef __attribute__((ext_vector_type(4)))  int       v4i;

#define IN_F   256
#define OUT_F  128
#define NB     4
#define NN     4096
#define NROWS  (NB * NN)   // 16384
#define LDA    40          // LDS row stride (bf16 elems): 64B row + 16B pad
#define NEG_SLOPE 0.01f

union FragU { uint4 q[2]; v16bf v; };

__device__ __forceinline__ void zero8(v8f& a) {
    for (int e = 0; e < 8; ++e) a[e] = 0.f;
}

// ---------------------------------------------------------------------------
// Kernel 0: W_fc [256][128] f32  ->  Wt [128][256] bf16 (B-operand layout)
// ---------------------------------------------------------------------------
__global__ __launch_bounds__(256)
void k_prep_w(const float* __restrict__ W, __bf16* __restrict__ Wt) {
    int e = blockIdx.x * 256 + threadIdx.x;        // 32768 elements
    int k = e >> 7, f = e & 127;
    Wt[(size_t)f * IN_F + k] = (__bf16)W[e];
}

// ---------------------------------------------------------------------------
// Kernel 1: feats = X @ W_fc  (bf16 WMMA, f32 accum)
//   grid: 128 blocks x 256 threads; block -> 128 rows x 128 cols
// ---------------------------------------------------------------------------
__global__ __launch_bounds__(256)
void k_feats(const float* __restrict__ X, const __bf16* __restrict__ Wt,
             __bf16* __restrict__ feats_rm, __bf16* __restrict__ featsT) {
    __shared__ __bf16 Atile[128 * LDA];

    const int tid  = threadIdx.x;
    const int lane = tid & 31, wave = tid >> 5;
    const int rowBase = blockIdx.x * 128;
    const int Moff = (wave & 3) * 32;              // 4 waves along M
    const int Noff = (wave >> 2) * 64;             // 2 waves along N
    const int r  = lane & 15;
    const int hi = lane >> 4;

    v8f acc[2][4];
    for (int m = 0; m < 2; ++m)
        for (int n = 0; n < 4; ++n) zero8(acc[m][n]);

    for (int k0 = 0; k0 < IN_F; k0 += 32) {
        __syncthreads();
        // stage A tile: 128 rows x 32 cols, f32 -> bf16 in LDS
        for (int s = 0; s < 4; ++s) {
            int q   = tid + 256 * s;               // float4 index, 1024 total
            int row = q >> 3, c4 = (q & 7) * 4;
            float4 xv = *(const float4*)(X + (size_t)(rowBase + row) * IN_F + k0 + c4);
            __bf16* dst = &Atile[row * LDA + c4];
            dst[0] = (__bf16)xv.x; dst[1] = (__bf16)xv.y;
            dst[2] = (__bf16)xv.z; dst[3] = (__bf16)xv.w;
        }
        __syncthreads();

        // A fragments from LDS: lane holds row M=r; K quads at hi*8 and hi*8+16
        FragU a[2];
        for (int m = 0; m < 2; ++m) {
            const __bf16* pa = &Atile[(Moff + m * 16 + r) * LDA + hi * 8];
            a[m].q[0] = *(const uint4*)pa;
            a[m].q[1] = *(const uint4*)(pa + 16);
        }
        // B fragments direct from global Wt (L2-resident): lane holds col N=r,
        // K quads at hi*8 and hi*8+16
        FragU b[4];
        for (int n = 0; n < 4; ++n) {
            const __bf16* pb = Wt + (size_t)(Noff + n * 16 + r) * IN_F + k0 + hi * 8;
            b[n].q[0] = *(const uint4*)pb;
            b[n].q[1] = *(const uint4*)(pb + 16);
        }
        for (int m = 0; m < 2; ++m)
            for (int n = 0; n < 4; ++n)
                acc[m][n] = __builtin_amdgcn_wmma_f32_16x16x32_bf16(
                    false, a[m].v, false, b[n].v, (short)0, acc[m][n], false, false);
    }

    // store: C/D layout -> row = hi*8 + e, col = lane&15
    for (int m = 0; m < 2; ++m)
        for (int n = 0; n < 4; ++n) {
            int colL = Noff + n * 16 + r;
            for (int e = 0; e < 8; ++e) {
                int grow = rowBase + Moff + m * 16 + hi * 8 + e;
                __bf16 h = (__bf16)acc[m][n][e];
                feats_rm[(size_t)grow * OUT_F + colL] = h;
                int bb = grow >> 12, j = grow & (NN - 1);
                featsT[((size_t)bb * OUT_F + colL) * NN + j] = h;
            }
        }
}

// ---------------------------------------------------------------------------
// Kernel 2: f1 = feats@w_l + b_l ; f2 = feats@w_r + b_r  (one wave per row)
// ---------------------------------------------------------------------------
__global__ __launch_bounds__(256)
void k_f1f2(const __bf16* __restrict__ feats_rm,
            const float* __restrict__ wl, const float* __restrict__ bl,
            const float* __restrict__ wr, const float* __restrict__ br,
            float* __restrict__ f1, float* __restrict__ f2) {
    int tid = threadIdx.x, lane = tid & 31, wave = tid >> 5;
    int row = blockIdx.x * 8 + wave;
    const __bf16* p = feats_rm + (size_t)row * OUT_F + lane * 4;
    float sl = 0.f, sr = 0.f;
    for (int e = 0; e < 4; ++e) {
        float v = (float)p[e];
        sl += v * wl[lane * 4 + e];
        sr += v * wr[lane * 4 + e];
    }
    for (int off = 16; off > 0; off >>= 1) {
        sl += __shfl_xor(sl, off, 32);
        sr += __shfl_xor(sr, off, 32);
    }
    if (lane == 0) {
        f1[row] = sl + bl[0];
        f2[row] = sr + br[0];
    }
}

// ---------------------------------------------------------------------------
// Kernel 3: softmax denominators over axis=1 (columns):
//   S[b][j] = sum_i exp(lrelu(f1[b][i] + f2[b][j]))
// ---------------------------------------------------------------------------
__global__ __launch_bounds__(256)
void k_colsum(const float* __restrict__ f1, const float* __restrict__ f2,
              float* __restrict__ S) {
    __shared__ float part[4][64];
    int tid = threadIdx.x;
    int jl = tid & 63, g = tid >> 6;
    int b  = blockIdx.x >> 6;
    int jt = blockIdx.x & 63;
    int j  = jt * 64 + jl;
    float f2j = f2[b * NN + j];
    const float* f1b = f1 + b * NN + g * 1024;
    float sum = 0.f;
    for (int i = 0; i < 1024; ++i) {
        float x = f1b[i] + f2j;                    // wave-uniform f1 load
        x = (x >= 0.f) ? x : NEG_SLOPE * x;
        sum += __expf(x);
    }
    part[g][jl] = sum;
    __syncthreads();
    if (tid < 64)
        S[b * NN + jt * 64 + tid] =
            part[0][tid] + part[1][tid] + part[2][tid] + part[3][tid];
}

// ---------------------------------------------------------------------------
// Kernel 4: out[b,i,:] = sum_j P[i,j] * feats[b,j,:]
//   P[i,j] = exp(lrelu(f1_i + f2_j)) / S_j, generated on-the-fly into LDS.
//   B tile (128f x 32j of featsT) is DMA'd into LDS by the Tensor Data Mover,
//   with TDM padding reproducing the LDA=40 layout; DMA overlaps the exp work.
// ---------------------------------------------------------------------------
__global__ __launch_bounds__(256)
void k_attn_out(const float* __restrict__ f1, const float* __restrict__ f2,
                const float* __restrict__ S, const __bf16* __restrict__ featsT,
                float* __restrict__ out) {
    __shared__ __bf16 Ptile[128 * LDA];
    __shared__ __bf16 Ft[128 * LDA];               // TDM destination
    __shared__ float  sf1[128];
    __shared__ float  f2s[32];
    __shared__ float  sinv[32];

    const int tid  = threadIdx.x;
    const int lane = tid & 31, wave = tid >> 5;
    const int b     = blockIdx.x >> 5;
    const int iTile = (blockIdx.x & 31) * 128;
    const int Moff = (wave & 3) * 32;
    const int Noff = (wave >> 2) * 64;
    const int r  = lane & 15;
    const int hi = lane >> 4;

    if (tid < 128) sf1[tid] = f1[b * NN + iTile + tid];

    // ---- TDM descriptor (2D, data_size=2B, tile 32 x 128, padded rows) ----
    // group1: d0 = mask0 | data_size(1)<<16 | pad_en<<20 | intv(3)<<22 | amt(3)<<25
    //         d1 = tensor_dim0.lo16 << 16 (=4096); d2 = tensor_dim1 << 16 (=128)
    //         d3 = tile_dim0 << 16 (=32);  d4 = tile_dim1 (=128)
    //         d5 = tensor_dim0_stride (=4096); d6 = d7 = 0
    const v8i g1 = { 0x06D10000, 0x10000000, 0x00800000, 0x00200000,
                     128, 4096, 0, 0 };
    const v4i gz4 = { 0, 0, 0, 0 };
    const v8i gz8 = { 0, 0, 0, 0, 0, 0, 0, 0 };
    const unsigned ldsOff = (unsigned)(size_t)(&Ft[0]);   // generic low 32 = LDS offset
    const unsigned long long gbase =
        (unsigned long long)(size_t)(featsT + (size_t)b * OUT_F * NN);

    v8f acc[2][4];
    for (int m = 0; m < 2; ++m)
        for (int n = 0; n < 4; ++n) zero8(acc[m][n]);

    const int iL = tid >> 1;                       // P-tile row handled by thread
    const int jb = (tid & 1) * 16;                 // P-tile col half

    for (int j0 = 0; j0 < NN; j0 += 32) {
        __syncthreads();                           // prev WMMA done reading LDS
        if (wave == 0) {
            unsigned long long ga = gbase + (unsigned long long)j0 * 2;
            v4u g0 = { 1u,                         // count=1 valid descriptor
                       ldsOff,
                       (unsigned)ga,
                       ((unsigned)(ga >> 32) & 0x1FFFFFFu) | (2u << 30) }; // type=2
            __builtin_amdgcn_tensor_load_to_lds(g0, g1, gz4, gz4, gz8, 0);
        }
        if (tid < 32) {
            f2s[tid]  = f2[b * NN + j0 + tid];
            sinv[tid] = 1.f / S[b * NN + j0 + tid];
        }
        __syncthreads();

        // build P tile [128 x 32] bf16 while the TDM streams the B tile
        {
            float f1i = sf1[iL];
            __bf16* dst = &Ptile[iL * LDA + jb];
            for (int jj = 0; jj < 16; ++jj) {
                float x = f1i + f2s[jb + jj];
                x = (x >= 0.f) ? x : NEG_SLOPE * x;
                dst[jj] = (__bf16)(__expf(x) * sinv[jb + jj]);
            }
        }
        if (wave == 0) __builtin_amdgcn_s_wait_tensorcnt(0);
        __syncthreads();                           // publish Ptile + TDM tile

        // A fragments (P) from LDS
        FragU a[2];
        for (int m = 0; m < 2; ++m) {
            const __bf16* pa = &Ptile[(Moff + m * 16 + r) * LDA + hi * 8];
            a[m].q[0] = *(const uint4*)pa;
            a[m].q[1] = *(const uint4*)(pa + 16);
        }
        // B fragments (feats^T tile) from LDS
        FragU bf[4];
        for (int n = 0; n < 4; ++n) {
            const __bf16* pb = &Ft[(Noff + n * 16 + r) * LDA + hi * 8];
            bf[n].q[0] = *(const uint4*)pb;
            bf[n].q[1] = *(const uint4*)(pb + 16);
        }
        for (int m = 0; m < 2; ++m)
            for (int n = 0; n < 4; ++n)
                acc[m][n] = __builtin_amdgcn_wmma_f32_16x16x32_bf16(
                    false, a[m].v, false, bf[n].v, (short)0, acc[m][n], false, false);
    }

    // write fp32 output
    for (int m = 0; m < 2; ++m)
        for (int n = 0; n < 4; ++n) {
            int colL = Noff + n * 16 + r;
            for (int e = 0; e < 8; ++e) {
                int i = iTile + Moff + m * 16 + hi * 8 + e;
                out[((size_t)b * NN + i) * OUT_F + colL] = acc[m][n][e];
            }
        }
}

// ---------------------------------------------------------------------------
extern "C" void kernel_launch(void* const* d_in, const int* in_sizes, int n_in,
                              void* d_out, int out_size, void* d_ws, size_t ws_size,
                              hipStream_t stream) {
    const float* X  = (const float*)d_in[0];   // [4,4096,256]
    const float* W  = (const float*)d_in[1];   // [256,128]
    const float* wl = (const float*)d_in[2];   // [128]
    const float* bl = (const float*)d_in[3];   // [1]
    const float* wr = (const float*)d_in[4];   // [128]
    const float* br = (const float*)d_in[5];   // [1]
    float* out = (float*)d_out;                // [4,4096,128] f32

    char* ws = (char*)d_ws;
    __bf16* Wt       = (__bf16*)ws; ws += (size_t)OUT_F * IN_F * 2;       //  64 KB
    __bf16* feats_rm = (__bf16*)ws; ws += (size_t)NROWS * OUT_F * 2;      // 4.2 MB
    __bf16* featsT   = (__bf16*)ws; ws += (size_t)NROWS * OUT_F * 2;      // 4.2 MB
    float*  f1       = (float*)ws;  ws += (size_t)NROWS * 4;              //  64 KB
    float*  f2v      = (float*)ws;  ws += (size_t)NROWS * 4;              //  64 KB
    float*  S        = (float*)ws;  ws += (size_t)NROWS * 4;              //  64 KB

    k_prep_w  <<<128,  256, 0, stream>>>(W, Wt);
    k_feats   <<<128,  256, 0, stream>>>(X, Wt, feats_rm, featsT);
    k_f1f2    <<<2048, 256, 0, stream>>>(feats_rm, wl, bl, wr, br, f1, f2v);
    k_colsum  <<<256,  256, 0, stream>>>(f1, f2v, S);
    k_attn_out<<<128,  256, 0, stream>>>(f1, f2v, S, featsT, out);
}
